// ScaledDotProductAttention_56066503082449
// MI455X (gfx1250) — matrix operand order repaired and checked by reading, not compiled, not run on hardware
//
#include <hip/hip_runtime.h>

// ---- problem constants (B,H,S,D from reference) ----
namespace {
constexpr int Bz = 2, Hz = 16, Sz = 2048, Dz = 64;
constexpr int KC  = 32;             // keys per chunk (two 16-wide WMMA sub-tiles)
constexpr int RPW = 32;             // query rows per wave (two 16-row WMMA tiles)
constexpr int RPB = 128;            // query rows per block (4 waves)
constexpr int KT_STRIDE = 40;       // bf16 per K^T row  -> [64][40]
constexpr int V_STRIDE  = 72;       // bf16 per V row    -> [32][72]
constexpr int P_STRIDE  = 40;       // bf16 per P row    -> [32][40] per wave
// Q pre-scale: (1/sqrt(64)) * log2(e), so scores are already in log2 domain
constexpr float QSCALE = 0.125f * 1.44269504088896340736f;
constexpr float MASKNEG = -1.5e9f;  // log2-domain "minus infinity" (exp2 -> 0)
}

typedef __attribute__((ext_vector_type(16))) __bf16        v16bf;
typedef __attribute__((ext_vector_type(8)))  float         v8f;
typedef __attribute__((ext_vector_type(4)))  unsigned int  u32x4;

union V16 { u32x4 q[2]; v16bf v; };
union BF2 { __bf16 h[2]; unsigned int u; };

__device__ __forceinline__ v16bf lds_v16(const __bf16* p) {
  V16 r;
  r.q[0] = *(const u32x4*)(p);      // 16B aligned by construction
  r.q[1] = *(const u32x4*)(p + 8);
  return r.v;
}

__global__ __launch_bounds__(128)
void sdpa_fwd_kernel(const float* __restrict__ Q, const float* __restrict__ K,
                     const float* __restrict__ V, const unsigned char* __restrict__ Mk,
                     float* __restrict__ out_ctx, float* __restrict__ out_prob)
{
  __shared__ __align__(16) __bf16        ktile[64 * KT_STRIDE];   // K^T bf16 [d][key]
  __shared__ __align__(16) __bf16        vtile[KC * V_STRIDE];    // V   bf16 [key][d]
  __shared__ __align__(16) __bf16        pbuf[4][RPW * P_STRIDE]; // P per wave
  __shared__ __align__(16) unsigned char mtile[RPB][KC];          // mask bytes

  const int tid  = threadIdx.x;
  const int lane = tid & 31;
  const int wv   = tid >> 5;
  const int n    = lane & 15;   // C/D column N = lane&15
  const int grp  = lane >> 4;   // half-wave group

  const int blk = blockIdx.x;
  const int qt  = blk & ((Sz / RPB) - 1);   // 0..15
  const int bh  = blk >> 4;                 // b*H + h
  const int b   = bh >> 4;

  const size_t ho   = (size_t)bh * Sz * Dz;
  const int    q0b  = qt * RPB;             // block's first query row
  const int    q0   = q0b + wv * RPW;       // this wave's first query row

  // staging roles: 4 threads per key row (K/V), 1 thread per mask row
  const int skl = tid >> 2;   // key_local 0..31
  const int st4 = tid & 3;    // quarter   0..3

  // ---- load Q in WMMA A layout (two 16-row sets), pre-scaled ----
  // A 16x32 bf16: lane row = lane&15; elems i<8 -> K=grp*8+i ; i>=8 -> K=16+grp*8+(i-8)
  V16 qa[4];  // [set*2 + c]
#pragma unroll
  for (int set = 0; set < 2; ++set) {
    const float* qp = Q + ho + (size_t)(q0 + set * 16 + n) * Dz;
#pragma unroll
    for (int c = 0; c < 2; ++c) {
      const int d0 = c * 32 + grp * 8;
      const int d1 = c * 32 + 16 + grp * 8;
      float4 f0 = *(const float4*)(qp + d0);
      float4 f1 = *(const float4*)(qp + d0 + 4);
      float4 f2 = *(const float4*)(qp + d1);
      float4 f3 = *(const float4*)(qp + d1 + 4);
      float tA[8] = { f0.x, f0.y, f0.z, f0.w, f1.x, f1.y, f1.z, f1.w };
      float tB[8] = { f2.x, f2.y, f2.z, f2.w, f3.x, f3.y, f3.z, f3.w };
      BF2 pk;
#pragma unroll
      for (int j = 0; j < 4; ++j) {
        pk.h[0] = (__bf16)(tA[2 * j] * QSCALE);
        pk.h[1] = (__bf16)(tA[2 * j + 1] * QSCALE);
        ((unsigned int*)&qa[set * 2 + c])[j] = pk.u;
        pk.h[0] = (__bf16)(tB[2 * j] * QSCALE);
        pk.h[1] = (__bf16)(tB[2 * j + 1] * QSCALE);
        ((unsigned int*)&qa[set * 2 + c])[4 + j] = pk.u;
      }
    }
  }

  // ---------------- pass 1: l[row] = sum_k exp2(masked score') --------------
  float l[16];
#pragma unroll
  for (int i = 0; i < 16; ++i) l[i] = 0.0f;

  for (int kb = 0; kb < Sz; kb += KC) {
    __syncthreads();
    { // stage K chunk transposed to LDS as bf16
      const float* kp = K + ho + (size_t)(kb + skl) * Dz + st4 * 16;
      float4 g0 = *(const float4*)(kp);
      float4 g1 = *(const float4*)(kp + 4);
      float4 g2 = *(const float4*)(kp + 8);
      float4 g3 = *(const float4*)(kp + 12);
      float tv[16] = { g0.x,g0.y,g0.z,g0.w, g1.x,g1.y,g1.z,g1.w,
                       g2.x,g2.y,g2.z,g2.w, g3.x,g3.y,g3.z,g3.w };
#pragma unroll
      for (int j = 0; j < 16; ++j)
        ktile[(st4 * 16 + j) * KT_STRIDE + skl] = (__bf16)tv[j];
    }
    { // stage mask bytes: one row (32B) per thread, coalesced b128
      const unsigned char* mp =
          Mk + (size_t)b * Sz * Sz + (size_t)(q0b + tid) * Sz + kb;
      *(u32x4*)&mtile[tid][0]  = *(const u32x4*)(mp);
      *(u32x4*)&mtile[tid][16] = *(const u32x4*)(mp + 16);
    }
    __syncthreads();

#pragma unroll
    for (int t = 0; t < 2; ++t) {
      v16bf b0, b1;
#pragma unroll
      for (int s = 0; s < 16; ++s) {
        const int kd = (s < 8) ? (grp * 8 + s) : (16 + grp * 8 + (s - 8));
        b0[s] = ktile[kd        * KT_STRIDE + t * 16 + n];
        b1[s] = ktile[(32 + kd) * KT_STRIDE + t * 16 + n];
      }
#pragma unroll
      for (int set = 0; set < 2; ++set) {
        v8f sc = {};
        sc = __builtin_amdgcn_wmma_f32_16x16x32_bf16(false, qa[set * 2 + 0].v, false, b0,
                                                     (short)0, sc, false, false);
        sc = __builtin_amdgcn_wmma_f32_16x16x32_bf16(false, qa[set * 2 + 1].v, false, b1,
                                                     (short)0, sc, false, false);
        const int lr = wv * RPW + set * 16 + 8 * grp;  // local mask row base
#pragma unroll
        for (int r = 0; r < 8; ++r) {
          // masked: s + mk*(-1.5e9) -> exp2 underflows to exactly 0
          float s = fmaf((float)mtile[lr + r][t * 16 + n], MASKNEG, sc[r]);
          l[set * 8 + r] += __builtin_amdgcn_exp2f(s);
        }
      }
    }
  }

  // sum partial l across the 16 lanes sharing each row; invert
#pragma unroll
  for (int i = 0; i < 16; ++i) {
#pragma unroll
    for (int sh = 1; sh < 16; sh <<= 1)
      l[i] += __shfl_xor(l[i], sh, 32);
    l[i] = 1.0f / l[i];   // l[] now holds 1/sum
  }

  // ---------------- pass 2: probs out + context = P @ V --------------------
  v8f acc[8] = {};   // [set*4 + dt]
  for (int kb = 0; kb < Sz; kb += KC) {
    __syncthreads();
    { // stage K^T
      const float* kp = K + ho + (size_t)(kb + skl) * Dz + st4 * 16;
      float4 g0 = *(const float4*)(kp);
      float4 g1 = *(const float4*)(kp + 4);
      float4 g2 = *(const float4*)(kp + 8);
      float4 g3 = *(const float4*)(kp + 12);
      float tv[16] = { g0.x,g0.y,g0.z,g0.w, g1.x,g1.y,g1.z,g1.w,
                       g2.x,g2.y,g2.z,g2.w, g3.x,g3.y,g3.z,g3.w };
#pragma unroll
      for (int j = 0; j < 16; ++j)
        ktile[(st4 * 16 + j) * KT_STRIDE + skl] = (__bf16)tv[j];
    }
    { // stage V row-major bf16 (packed b32 stores)
      const float* vp = V + ho + (size_t)(kb + skl) * Dz + st4 * 16;
      float4 g0 = *(const float4*)(vp);
      float4 g1 = *(const float4*)(vp + 4);
      float4 g2 = *(const float4*)(vp + 8);
      float4 g3 = *(const float4*)(vp + 12);
      float tv[16] = { g0.x,g0.y,g0.z,g0.w, g1.x,g1.y,g1.z,g1.w,
                       g2.x,g2.y,g2.z,g2.w, g3.x,g3.y,g3.z,g3.w };
#pragma unroll
      for (int j = 0; j < 8; ++j) {
        BF2 pk;
        pk.h[0] = (__bf16)tv[2 * j];
        pk.h[1] = (__bf16)tv[2 * j + 1];
        *(unsigned int*)&vtile[skl * V_STRIDE + st4 * 16 + 2 * j] = pk.u;
      }
    }
    { // stage mask bytes
      const unsigned char* mp =
          Mk + (size_t)b * Sz * Sz + (size_t)(q0b + tid) * Sz + kb;
      *(u32x4*)&mtile[tid][0]  = *(const u32x4*)(mp);
      *(u32x4*)&mtile[tid][16] = *(const u32x4*)(mp + 16);
    }
    __syncthreads();

#pragma unroll
    for (int t = 0; t < 2; ++t) {
      v16bf b0, b1;
#pragma unroll
      for (int s = 0; s < 16; ++s) {
        const int kd = (s < 8) ? (grp * 8 + s) : (16 + grp * 8 + (s - 8));
        b0[s] = ktile[kd        * KT_STRIDE + t * 16 + n];
        b1[s] = ktile[(32 + kd) * KT_STRIDE + t * 16 + n];
      }
      const int key = kb + t * 16 + n;
#pragma unroll
      for (int set = 0; set < 2; ++set) {
        v8f sc = {};
        sc = __builtin_amdgcn_wmma_f32_16x16x32_bf16(false, qa[set * 2 + 0].v, false, b0,
                                                     (short)0, sc, false, false);
        sc = __builtin_amdgcn_wmma_f32_16x16x32_bf16(false, qa[set * 2 + 1].v, false, b1,
                                                     (short)0, sc, false, false);
        const int lr = wv * RPW + set * 16 + 8 * grp;
        float* pp = out_prob + (size_t)bh * Sz * Sz +
                    (size_t)(q0 + set * 16 + 8 * grp) * Sz + key;
#pragma unroll
        for (int r = 0; r < 8; ++r) {
          float s = fmaf((float)mtile[lr + r][t * 16 + n], MASKNEG, sc[r]);
          float p = __builtin_amdgcn_exp2f(s) * l[set * 8 + r];  // normalized
          __builtin_nontemporal_store(p, pp + (size_t)r * Sz);   // streaming f32
          pbuf[wv][(set * 16 + r + 8 * grp) * P_STRIDE + t * 16 + n] = (__bf16)p;
        }
      }
    }

    asm volatile("s_wait_dscnt 0" ::: "memory");   // order P store -> P load

    { // P as WMMA-A (16x32) per set, V chunks as WMMA-B, accumulate context
      V16 paA, paB;
      const __bf16* ppA = &pbuf[wv][(0  + n) * P_STRIDE + grp * 8];
      const __bf16* ppB = &pbuf[wv][(16 + n) * P_STRIDE + grp * 8];
      paA.q[0] = *(const u32x4*)(ppA);
      paA.q[1] = *(const u32x4*)(ppA + 16);
      paB.q[0] = *(const u32x4*)(ppB);
      paB.q[1] = *(const u32x4*)(ppB + 16);
#pragma unroll
      for (int dt = 0; dt < 4; ++dt) {
        v16bf bv;
#pragma unroll
        for (int s = 0; s < 16; ++s) {
          const int kk = (s < 8) ? (grp * 8 + s) : (16 + grp * 8 + (s - 8));
          bv[s] = vtile[kk * V_STRIDE + dt * 16 + n];
        }
        acc[dt]     = __builtin_amdgcn_wmma_f32_16x16x32_bf16(false, paA.v, false, bv,
                                                              (short)0, acc[dt],
                                                              false, false);
        acc[4 + dt] = __builtin_amdgcn_wmma_f32_16x16x32_bf16(false, paB.v, false, bv,
                                                              (short)0, acc[4 + dt],
                                                              false, false);
      }
    }
  }

  // ---------------- write context ------------------------------------------
#pragma unroll
  for (int set = 0; set < 2; ++set) {
    float* cp = out_ctx + ho + (size_t)(q0 + set * 16 + 8 * grp) * Dz + n;
#pragma unroll
    for (int dt = 0; dt < 4; ++dt) {
#pragma unroll
      for (int r = 0; r < 8; ++r)
        __builtin_nontemporal_store(acc[set * 4 + dt][r], cp + r * Dz + dt * 16);
    }
  }
}

extern "C" void kernel_launch(void* const* d_in, const int* in_sizes, int n_in,
                              void* d_out, int out_size, void* d_ws, size_t ws_size,
                              hipStream_t stream) {
  (void)in_sizes; (void)n_in; (void)out_size; (void)d_ws; (void)ws_size;
  const float*         Q  = (const float*)d_in[0];
  const float*         K  = (const float*)d_in[1];
  const float*         V  = (const float*)d_in[2];
  const unsigned char* Mk = (const unsigned char*)d_in[3]; // jnp.bool_ -> 1 byte

  float* ctx  = (float*)d_out;                              // (B,H,S,D) first
  float* prob = ctx + (size_t)Bz * Hz * Sz * Dz;            // then (B,H,S,S)

  dim3 grid(Bz * Hz * (Sz / RPB));  // 512 blocks
  dim3 block(128);                  // 4 waves of 32
  hipLaunchKernelGGL(sdpa_fwd_kernel, grid, block, 0, stream, Q, K, V, Mk, ctx, prob);
}
